// AttentionLayer_86234353369960
// MI455X (gfx1250) — matrix operand order repaired and checked by reading, not compiled, not run on hardware
//
#include <hip/hip_runtime.h>
#include <hip/hip_bf16.h>

typedef __attribute__((ext_vector_type(16))) _Float16 v16h;
typedef __attribute__((ext_vector_type(8)))  _Float16 v8h;
typedef __attribute__((ext_vector_type(8)))  float    v8f;
typedef __attribute__((ext_vector_type(4)))  int      v4i;

#define B_  8
#define L_  4096
#define C_  256
#define D_  128
#define BL_ 32768   // B_*L_
#define KC_ 64      // keys per flash chunk

// workspace offsets (in _Float16 elements)
#define OFF_XH 0ull
#define OFF_QH 8388608ull                 // BL_*C_
#define OFF_KH (OFF_QH + 4194304ull)      // + BL_*D_
#define OFF_VT (OFF_KH + 4194304ull)
#define OFF_AH (OFF_VT + 4194304ull)
#define OFF_WQ (OFF_AH + 4194304ull)
#define OFF_WK (OFF_WQ + 32768ull)
#define OFF_WV (OFF_WK + 32768ull)
#define OFF_WL (OFF_WV + 32768ull)

#if __has_builtin(__builtin_amdgcn_global_load_async_to_lds_b128)
#define HAVE_ASYNC_LDS 1
#else
#define HAVE_ASYNC_LDS 0
#endif

typedef __attribute__((address_space(1))) v4i g_v4i;
typedef __attribute__((address_space(3))) v4i l_v4i;

// 16-byte global -> LDS copy; async (ASYNCcnt-tracked) when available.
__device__ __forceinline__ void async_cp16(_Float16* dst_lds, const _Float16* src_global) {
#if HAVE_ASYNC_LDS
  __builtin_amdgcn_global_load_async_to_lds_b128(
      (g_v4i*)src_global, (l_v4i*)dst_lds, 0, 0);
#else
  *(v8h*)dst_lds = *(const v8h*)src_global;
#endif
}

__device__ __forceinline__ void wait_async0() {
#if HAVE_ASYNC_LDS
#if __has_builtin(__builtin_amdgcn_s_wait_asynccnt)
  __builtin_amdgcn_s_wait_asynccnt(0);
#else
  asm volatile("s_wait_asynccnt 0x0" ::: "memory");
#endif
#endif
}

__device__ __forceinline__ v8f wmma16(v16h a, v16h b, v8f c) {
  return __builtin_amdgcn_wmma_f32_16x16x32_f16(false, a, false, b, (short)0, c,
                                                false, false);
}

// A fragment (16x32 f16, MxK), source row-major, leading dim `ld`, p -> [row0][k0].
__device__ __forceinline__ v16h load_a16(const _Float16* p, int ld, int lane) {
  const int r  = lane & 15;
  const int hi = lane >> 4;
  const _Float16* b0 = p + (size_t)r * ld + hi * 16;
  v16h a;
#pragma unroll
  for (int i = 0; i < 16; ++i) a[i] = b0[i];
  return a;
}

// B fragment (32x16 f16, KxN) from TRANSPOSED storage M[n][k] (row-major, ld = k-stride).
__device__ __forceinline__ v16h load_b16(const _Float16* p, int ld, int lane) {
  const int n  = lane & 15;
  const int kb = (lane >> 4) * 16;
  const _Float16* b0 = p + (size_t)n * ld + kb;
  v16h b;
#pragma unroll
  for (int i = 0; i < 16; ++i) b[i] = b0[i];
  return b;
}

// ---------------------------------------------------------------------------
// Kernel 1: f32 -> f16 conversion; weights stored transposed.
// ---------------------------------------------------------------------------
__global__ void cvt_kernel(const float* __restrict__ x,  const float* __restrict__ Wq,
                           const float* __restrict__ Wk, const float* __restrict__ Wv,
                           const float* __restrict__ Wl, _Float16* __restrict__ ws) {
  size_t i = (size_t)blockIdx.x * blockDim.x + threadIdx.x;
  const size_t NX = (size_t)BL_ * C_;
  if (i < NX) { ws[OFF_XH + i] = (_Float16)x[i]; return; }
  size_t j = i - NX;
  if (j >= 4ull * 32768ull) return;
  const int which = (int)(j >> 15);
  const int t     = (int)(j & 32767);
  if (which < 3) {
    const float* W = (which == 0) ? Wq : ((which == 1) ? Wk : Wv);
    _Float16* wt = ws + ((which == 0) ? OFF_WQ : ((which == 1) ? OFF_WK : OFF_WV));
    const int cin = t >> 7, d = t & 127;
    wt[(size_t)d * C_ + cin] = (_Float16)W[t];        // [D][C]
  } else {
    const int din = t >> 8, cout = t & 255;
    ws[OFF_WL + (size_t)cout * D_ + din] = (_Float16)Wl[t];  // [C][D]
  }
}

// ---------------------------------------------------------------------------
// Kernel 2: q/k/v projections.  Weights staged in LDS (64 KB), A fragments
// preloaded, inner loop = ds_load + wmma only.
// ---------------------------------------------------------------------------
__global__ void __launch_bounds__(256) proj_kernel(_Float16* __restrict__ ws) {
  __shared__ __attribute__((aligned(16))) _Float16 Wlds[D_ * C_];  // 64 KB
  const int tid  = threadIdx.x;
  const int lane = tid & 31;
  const int wave = tid >> 5;
  const int n  = lane & 15;
  const int hi = lane >> 4;
  const int row0 = (blockIdx.x * 8 + wave) * 16;
  const int proj = blockIdx.y;
  const _Float16* xh = ws + OFF_XH;
  const _Float16* wt = ws + ((proj == 0) ? OFF_WQ : ((proj == 1) ? OFF_WK : OFF_WV));

#pragma unroll
  for (int i = 0; i < 16; ++i)
    async_cp16(Wlds + (size_t)(tid + 256 * i) * 8, wt + (size_t)(tid + 256 * i) * 8);

  v16h af[8];
#pragma unroll
  for (int c = 0; c < 8; ++c)
    af[c] = load_a16(xh + (size_t)row0 * C_ + 32 * c, C_, lane);

  wait_async0();
  __syncthreads();

  v8f acc[8] = {};
#pragma unroll
  for (int c = 0; c < 8; ++c) {
    v16h barr[8];
#pragma unroll
    for (int s = 0; s < 8; ++s)
      barr[s] = load_b16(Wlds + (size_t)(16 * s) * C_ + 32 * c, C_, lane);
#pragma unroll
    for (int s = 0; s < 8; ++s) acc[s] = wmma16(af[c], barr[s], acc[s]);
  }

  if (proj == 2) {                            // V: store transposed [b][d][l]
    const int bidx = row0 >> 12;
    const int l0   = row0 & (L_ - 1);
    _Float16* vt = ws + OFF_VT + (size_t)bidx * D_ * L_;
#pragma unroll
    for (int s = 0; s < 8; ++s) {
      v8h pk;
#pragma unroll
      for (int j = 0; j < 8; ++j) pk[j] = (_Float16)acc[s][j];
      *(v8h*)(vt + (size_t)(16 * s + n) * L_ + l0 + 8 * hi) = pk;
    }
  } else {                                    // Q/K: row-major [row][d]
    _Float16* dst = ws + ((proj == 0) ? OFF_QH : OFF_KH);
#pragma unroll
    for (int s = 0; s < 8; ++s)
#pragma unroll
      for (int j = 0; j < 8; ++j)
        dst[(size_t)(row0 + j + 8 * hi) * D_ + 16 * s + n] = (_Float16)acc[s][j];
  }
}

// ---------------------------------------------------------------------------
// Kernel 3: fused flash attention, 64-key chunks, double-buffered async K/V
// staging, shifted-softmax, depth-2 software-pipelined WMMA chains.
// ---------------------------------------------------------------------------
__global__ void __launch_bounds__(256) flash_kernel(_Float16* __restrict__ ws) {
  __shared__ __attribute__((aligned(16))) _Float16 Kbuf[2][KC_][D_];  // 2 x 16 KB
  __shared__ __attribute__((aligned(16))) _Float16 Vbuf[2][D_][KC_];  // 2 x 16 KB
  __shared__ __attribute__((aligned(16))) _Float16 ldsP[8][16][KC_];  // 16 KB
  const int tid  = threadIdx.x;
  const int lane = tid & 31;
  const int wave = tid >> 5;
  const int n  = lane & 15;
  const int hi = lane >> 4;
  const int gRow0 = blockIdx.x * 128 + wave * 16;
  const int bidx  = gRow0 >> 12;
  const _Float16* qh = ws + OFF_QH;
  const _Float16* kh = ws + OFF_KH + (size_t)bidx * L_ * D_;
  const _Float16* vt = ws + OFF_VT + (size_t)bidx * D_ * L_;
  _Float16* ah = ws + OFF_AH;

  v16h qf[4];
#pragma unroll
  for (int c = 0; c < 4; ++c)
    qf[c] = load_a16(qh + (size_t)gRow0 * D_ + 32 * c, D_, lane);

  v8f O[8] = {};
  float sacc[8] = {};          // per-lane partial sum of exp (keys == n mod 16)

  // cooperative stage of one 64-key chunk (K: 16 KB contiguous; V: 128 x 128 B)
  auto stage = [&](int key0, int bufi) {
    const _Float16* ksrc = kh + (size_t)key0 * D_;
#pragma unroll
    for (int i = 0; i < 4; ++i) {
      const int id = tid + 256 * i;           // 0..1023 x 16B
      async_cp16(&Kbuf[bufi][0][0] + (size_t)id * 8, ksrc + (size_t)id * 8);
    }
#pragma unroll
    for (int i = 0; i < 4; ++i) {
      const int id = tid + 256 * i;           // 0..1023
      const int d = id >> 3, part = id & 7;   // 8 x 16B per 64-half d-row
      async_cp16(&Vbuf[bufi][d][part * 8], vt + (size_t)d * L_ + key0 + part * 8);
    }
  };

  stage(0, 0);

  for (int key0 = 0; key0 < L_; key0 += KC_) {
    const int bufi = (key0 / KC_) & 1;
    wait_async0();          // my chunk-key0 transfers complete
    __syncthreads();        // everyone's transfers complete + prev compute done
    if (key0 + KC_ < L_) stage(key0 + KC_, bufi ^ 1);

    // ---- S = Q * K^T : four 16x16 C tiles; depth-2 pipelined K fragments ----
    v8f S[4] = {};
    v16h kq0 = load_b16(&Kbuf[bufi][0][0], D_, lane);          // (t=0,c=0)
    v16h kq1 = load_b16(&Kbuf[bufi][0][0] + 32, D_, lane);     // (t=0,c=1)
#pragma unroll
    for (int idx = 0; idx < 16; ++idx) {
      const int t = idx >> 2, c = idx & 3;
      const int ni = (idx < 14) ? idx + 2 : 15;                // lookahead 2
      const int nt = ni >> 2, nc = ni & 3;
      v16h kn = load_b16(&Kbuf[bufi][16 * nt][0] + 32 * nc, D_, lane);
      S[t] = wmma16(qf[c], kq0, S[t]);
      kq0 = kq1; kq1 = kn;
    }
    // ---- shifted exp; accumulate per-lane partial sums; write P tile ----
#pragma unroll
    for (int t = 0; t < 4; ++t)
#pragma unroll
      for (int j = 0; j < 8; ++j) {
        const float e = __expf(fminf(S[t][j] - 20.0f, 11.0f));  // f16-safe
        sacc[j] += e;
        ldsP[wave][j + 8 * hi][16 * t + n] = (_Float16)e;
      }
    asm volatile("s_wait_dscnt 0x0" ::: "memory");
    // ---- O += P * V (two 32-key sub-chunks); depth-2 pipelined V frags ----
#pragma unroll
    for (int kk = 0; kk < 2; ++kk) {
      v16h pf  = load_a16(&ldsP[wave][0][32 * kk], KC_, lane);
      v16h vq0 = load_b16(&Vbuf[bufi][0][32 * kk],  KC_, lane);
      v16h vq1 = load_b16(&Vbuf[bufi][16][32 * kk], KC_, lane);
#pragma unroll
      for (int s = 0; s < 8; ++s) {
        const int ns = (s < 6) ? s + 2 : 7;                    // lookahead 2
        v16h vn = load_b16(&Vbuf[bufi][16 * ns][32 * kk], KC_, lane);
        O[s] = wmma16(pf, vq0, O[s]);
        vq0 = vq1; vq1 = vn;
      }
    }
  }

  // ---- one final cross-lane reduction of the softmax denominators ----
#pragma unroll
  for (int j = 0; j < 8; ++j) {
    float t = sacc[j];
#pragma unroll
    for (int w = 1; w < 16; w <<= 1) t += __shfl_xor(t, w, 32);
    const float inv = 1.0f / t;
#pragma unroll
    for (int s = 0; s < 8; ++s)
      ah[(size_t)(gRow0 + j + 8 * hi) * D_ + 16 * s + n] =
          (_Float16)(O[s][j] * inv);
  }
}

// ---------------------------------------------------------------------------
// Kernel 4: out = gamma * (attn @ Wlast) + x.  Wlast^T staged in LDS.
// ---------------------------------------------------------------------------
__global__ void __launch_bounds__(256) reproj_kernel(const float* __restrict__ x,
                                                     const float* __restrict__ gamma,
                                                     float* __restrict__ out,
                                                     const _Float16* __restrict__ ws) {
  __shared__ __attribute__((aligned(16))) _Float16 Wlds[C_ * D_];  // 64 KB
  const int tid  = threadIdx.x;
  const int lane = tid & 31;
  const int wave = tid >> 5;
  const int n  = lane & 15;
  const int hi = lane >> 4;
  const int row0 = (blockIdx.x * 8 + wave) * 16;
  const _Float16* ah  = ws + OFF_AH;
  const _Float16* wlt = ws + OFF_WL;

#pragma unroll
  for (int i = 0; i < 16; ++i)
    async_cp16(Wlds + (size_t)(tid + 256 * i) * 8,
               (const _Float16*)wlt + (size_t)(tid + 256 * i) * 8);

  v16h af[4];
#pragma unroll
  for (int c = 0; c < 4; ++c)
    af[c] = load_a16(ah + (size_t)row0 * D_ + 32 * c, D_, lane);

  wait_async0();
  __syncthreads();

  v8f acc[16] = {};
#pragma unroll
  for (int c = 0; c < 4; ++c) {
#pragma unroll
    for (int s = 0; s < 16; ++s) {
      v16h b = load_b16(Wlds + (size_t)(16 * s) * D_ + 32 * c, D_, lane);
      acc[s] = wmma16(af[c], b, acc[s]);
    }
  }
  const float g = gamma[0];
#pragma unroll
  for (int s = 0; s < 16; ++s)
#pragma unroll
    for (int j = 0; j < 8; ++j) {
      const size_t idx = (size_t)(row0 + j + 8 * hi) * C_ + 16 * s + n;
      out[idx] = g * acc[s][j] + x[idx];
    }
}

extern "C" void kernel_launch(void* const* d_in, const int* in_sizes, int n_in,
                              void* d_out, int out_size, void* d_ws, size_t ws_size,
                              hipStream_t stream) {
  (void)in_sizes; (void)n_in; (void)out_size; (void)ws_size;
  const float* x     = (const float*)d_in[0];
  const float* Wq    = (const float*)d_in[1];
  const float* Wk    = (const float*)d_in[2];
  const float* Wv    = (const float*)d_in[3];
  const float* Wl    = (const float*)d_in[4];
  const float* gamma = (const float*)d_in[5];
  float*    out = (float*)d_out;
  _Float16* ws  = (_Float16*)d_ws;

  const size_t n_cvt = (size_t)BL_ * C_ + 4ull * 32768ull;
  const int cvt_blocks = (int)((n_cvt + 255) / 256);
  cvt_kernel<<<cvt_blocks, 256, 0, stream>>>(x, Wq, Wk, Wv, Wl, ws);
  proj_kernel<<<dim3(BL_ / (16 * 8), 3), 256, 0, stream>>>(ws);
  flash_kernel<<<BL_ / 128, 256, 0, stream>>>(ws);
  reproj_kernel<<<BL_ / (16 * 8), 256, 0, stream>>>(x, gamma, out, ws);
}
